// EGCLayer_26147760898656
// MI455X (gfx1250) — matrix operand order repaired and checked by reading, not compiled, not run on hardware
//
#include <hip/hip_runtime.h>
#include <cmath>

typedef __attribute__((ext_vector_type(16))) _Float16 v16h;
typedef __attribute__((ext_vector_type(8)))  float    v8f;

#define BB 4
#define NN 256
#define FF 128
#define EPS_LN 1e-5f

__device__ __forceinline__ float fastrcp(float v) { return __builtin_amdgcn_rcpf(v); }
__device__ __forceinline__ float siluf(float v) { return v * fastrcp(1.0f + __expf(-v)); }
__device__ __forceinline__ float sigmf(float v) { return fastrcp(1.0f + __expf(-v)); }

// Async DMA: global -> LDS, 16 bytes per lane, tracked by ASYNCcnt.
// (GLOBAL_LOAD_ASYNC_TO_LDS_B128, CDNA5 ISA ch.10 / 15.18.3)
__device__ __forceinline__ void async_copy_b128(unsigned lds_off, const void* gaddr) {
  asm volatile("global_load_async_to_lds_b128 %0, %1, off"
               :: "v"(lds_off), "v"(gaddr) : "memory");
}
__device__ __forceinline__ void wait_async0() {
  asm volatile("s_wait_asynccnt 0" ::: "memory");
}

// ---------------------------------------------------------------------------
// WMMA fragment loaders (gfx1250 wave32 layouts per CDNA5 ISA 7.12.2)
// A (16x32 f16, M x K): lanes 0-15 -> M=lane, K = {0..7,16..23}; lanes 16-31 ->
//   M=lane-16, K = {8..15,24..31}; 2 halves per VGPR.
__device__ __forceinline__ v16h load_a_frag(const _Float16* base, int ld) {
  int lane = threadIdx.x & 31;
  int m = lane & 15;
  int hi = lane >> 4;
  const _Float16* row = base + m * ld + 8 * hi;
  v16h a;
#pragma unroll
  for (int r = 0; r < 4; ++r) {
    a[2 * r]     = row[2 * r];
    a[2 * r + 1] = row[2 * r + 1];
    a[2 * r + 8] = row[16 + 2 * r];
    a[2 * r + 9] = row[16 + 2 * r + 1];
  }
  return a;
}

// B (32x16 f16, K x N) from weight W (Fout x Fin row-major f16 in LDS):
// B[k,n] = W[fo0+n, fi0+k]. lanes 0-15: n=lane, K={0..7,16..23}; lanes 16-31: n=lane-16,
// K={8..15,24..31} (matches A fragment order).
__device__ __forceinline__ v16h load_b_frag(const _Float16* w, int fo0, int fi0) {
  int lane = threadIdx.x & 31;
  int n = lane & 15;
  int hi = lane >> 4;
  const _Float16* p = w + (fo0 + n) * FF + fi0 + 8 * hi;
  v16h b;
#pragma unroll
  for (int r = 0; r < 16; ++r) b[r] = p[r < 8 ? r : r + 8];
  return b;
}

// ---------------------------------------------------------------------------
// Stage 0: one-shot f32 -> f16 conversion of the two 128x128 layer-2 weight
// matrices into workspace (so stage2 can DMA them straight into LDS).
__global__ void __launch_bounds__(256) stage0_cvt_weights(
    const float* __restrict__ edg2_w, const float* __restrict__ cor2_w,
    _Float16* __restrict__ w16) {
  int idx = blockIdx.x * 256 + threadIdx.x;  // 0 .. 2*FF*FF-1
  float v = (idx < FF * FF) ? edg2_w[idx] : cor2_w[idx - FF * FF];
  w16[idx] = (_Float16)v;
}

// ---------------------------------------------------------------------------
// Stage 1: LayerNorm h & x, per-node projections through the split first
// pair-layer weights, and zero-init of the global accumulators.
__global__ void __launch_bounds__(128) stage1_ln_proj(
    const float* __restrict__ x, const float* __restrict__ h,
    const float* __restrict__ lnh_w, const float* __restrict__ lnh_b,
    const float* __restrict__ lnx_w, const float* __restrict__ lnx_b,
    const float* __restrict__ edg1_w, const float* __restrict__ cor1_w,
    float* __restrict__ hln, float* __restrict__ xln,
    float* __restrict__ Ae, float* __restrict__ Be,
    float* __restrict__ Ac, float* __restrict__ Bc,
    float* __restrict__ aggG, float* __restrict__ xsumG) {
  int row = blockIdx.x;  // b*N + i
  int t = threadIdx.x;   // 0..127
  __shared__ float sh[FF];
  __shared__ float red[128];

  aggG[row * FF + t] = 0.f;
  if (t < 3) xsumG[row * 3 + t] = 0.f;

  float v = h[row * FF + t];
  red[t] = v;
  __syncthreads();
  for (int s = 64; s > 0; s >>= 1) { if (t < s) red[t] += red[t + s]; __syncthreads(); }
  float mean = red[0] * (1.0f / FF);
  __syncthreads();
  float d = v - mean;
  red[t] = d * d;
  __syncthreads();
  for (int s = 64; s > 0; s >>= 1) { if (t < s) red[t] += red[t + s]; __syncthreads(); }
  float var = red[0] * (1.0f / FF);
  float hl = d * rsqrtf(var + EPS_LN) * lnh_w[t] + lnh_b[t];
  sh[t] = hl;
  hln[row * FF + t] = hl;

  if (t == 0) {
    float a0 = x[row * 3 + 0], a1 = x[row * 3 + 1], a2 = x[row * 3 + 2];
    float m = (a0 + a1 + a2) * (1.0f / 3.0f);
    float vv = ((a0 - m) * (a0 - m) + (a1 - m) * (a1 - m) + (a2 - m) * (a2 - m)) * (1.0f / 3.0f);
    float inv = rsqrtf(vv + EPS_LN);
    xln[row * 3 + 0] = (a0 - m) * inv * lnx_w[0] + lnx_b[0];
    xln[row * 3 + 1] = (a1 - m) * inv * lnx_w[1] + lnx_b[1];
    xln[row * 3 + 2] = (a2 - m) * inv * lnx_w[2] + lnx_b[2];
  }
  __syncthreads();

  float se = 0.f, sbe = 0.f, sc = 0.f, sbc = 0.f;
  const float* we = edg1_w + t * (2 * FF + 2);
  const float* wc = cor1_w + t * (2 * FF + 2);
#pragma unroll 8
  for (int k = 0; k < FF; ++k) {
    float hv = sh[k];
    se  += hv * we[k];
    sbe += hv * we[FF + k];
    sc  += hv * wc[k];
    sbc += hv * wc[FF + k];
  }
  Ae[row * FF + t] = se;
  Be[row * FF + t] = sbe;
  Ac[row * FF + t] = sc;
  Bc[row * FF + t] = sbc;
}

// ---------------------------------------------------------------------------
// Stage 2: pairwise edge + coordinate MLPs with WMMA second layers.
// Block = (batch b, 16-row i tile, j-split); 4 waves; wave w owns j-tile
// jsplit*4 + w.  Per (i, j-tile) unit: M=16 pair rows (fixed i, 16 j's).
// Partial sums accumulated to global via atomic f32 adds (no-return).
#define SMEM_F32_COUNT (10 * FF + 4 * 64 + 4 * 16 * FF)
#define SMEM_F16_COUNT (FF * FF + FF * FF + 4 * 16 * FF)
#define SMEM_BYTES (SMEM_F32_COUNT * 4 + SMEM_F16_COUNT * 2)

__global__ void __launch_bounds__(128) stage2_pairs(
    const float* __restrict__ x, const float* __restrict__ x0,
    const float* __restrict__ Ae, const float* __restrict__ Be,
    const float* __restrict__ Ac, const float* __restrict__ Bc,
    const _Float16* __restrict__ w16,  // [edg2_w | cor2_w] as f16
    const float* __restrict__ edg1_w, const float* __restrict__ edg1_b,
    const float* __restrict__ edg2_b,
    const float* __restrict__ edgi_w, const float* __restrict__ edgi_b,
    const float* __restrict__ cor1_w, const float* __restrict__ cor1_b,
    const float* __restrict__ cor2_b,
    const float* __restrict__ cor3_w, const float* __restrict__ cor3_b,
    float* __restrict__ aggG, float* __restrict__ xsumG) {
  extern __shared__ char smemRaw[];
  float* wdE = (float*)smemRaw;
  float* wrE = wdE + FF;
  float* bE  = wrE + FF;
  float* b2E = bE + FF;
  float* wiE = b2E + FF;
  float* wdC = wiE + FF;
  float* wrC = wdC + FF;
  float* bC  = wrC + FF;
  float* b2C = bC + FF;
  float* w3C = b2C + FF;
  float* scS = w3C + FF;       // 4 waves * 64
  float* o2S = scS + 4 * 64;   // 4 waves * 16*FF
  _Float16* w2e = (_Float16*)(o2S + 4 * 16 * FF);
  _Float16* w2c = w2e + FF * FF;
  _Float16* m1S = w2c + FF * FF;  // 4 waves * 16*FF

  int jsplit = blockIdx.x & 3;
  int i0 = ((blockIdx.x >> 2) & 15) << 4;
  int b = blockIdx.x >> 6;
  int tid = threadIdx.x;
  int wave = tid >> 5;
  int lane = tid & 31;
  int p = lane & 15;            // pair row within tile (= j offset)
  int hi = lane >> 4;
  int fhalf = hi * 64;          // each lane handles 64 feature values

  // DMA both f16 weight matrices (64KB) straight into LDS, bypassing VGPRs.
  {
    unsigned ldsBase = (unsigned)(unsigned long long)(uintptr_t)w2e;
    const char* g = (const char*)w16;
#pragma unroll
    for (int c = 0; c < (2 * FF * FF * 2) / (16 * 128); ++c) {  // 32 chunks/thread
      int idx = c * 128 + tid;
      async_copy_b128(ldsBase + idx * 16, g + idx * 16);
    }
  }
  for (int idx = tid; idx < FF; idx += 128) {
    wdE[idx] = edg1_w[idx * (2 * FF + 2) + 2 * FF];
    wrE[idx] = edg1_w[idx * (2 * FF + 2) + 2 * FF + 1];
    bE[idx]  = edg1_b[idx];
    b2E[idx] = edg2_b[idx];
    wiE[idx] = edgi_w[idx];
    wdC[idx] = cor1_w[idx * (2 * FF + 2) + 2 * FF];
    wrC[idx] = cor1_w[idx * (2 * FF + 2) + 2 * FF + 1];
    bC[idx]  = cor1_b[idx];
    b2C[idx] = cor2_b[idx];
    w3C[idx] = cor3_w[idx];
  }
  wait_async0();
  __syncthreads();

  float eib = edgi_b[0];
  float c3b = cor3_b[0];

  _Float16* myM1 = m1S + wave * (16 * FF);
  float* myO2 = o2S + wave * (16 * FF);
  float* mySc = scS + wave * 64;

  int jt = jsplit * 4 + wave;   // this wave's j-tile
  int jb = jt * 16;
  int j = jb + p;
  const float* xj = x + (b * NN + j) * 3;
  const float* yj = x0 + (b * NN + j) * 3;
  float xj0 = xj[0], xj1 = xj[1], xj2 = xj[2];
  float yj0 = yj[0], yj1 = yj[1], yj2 = yj[2];
  const float* BeJ = Be + (b * NN + j) * FF;
  const float* BcJ = Bc + (b * NN + j) * FF;
  int nsel = lane & 15;

  for (int it = 0; it < 16; ++it) {
    int i = i0 + it;
    const float* AeRow = Ae + (b * NN + i) * FF;
    const float* AcRow = Ac + (b * NN + i) * FF;
    float xi0 = x[(b * NN + i) * 3 + 0];
    float xi1 = x[(b * NN + i) * 3 + 1];
    float xi2 = x[(b * NN + i) * 3 + 2];
    float yi0 = x0[(b * NN + i) * 3 + 0];
    float yi1 = x0[(b * NN + i) * 3 + 1];
    float yi2 = x0[(b * NN + i) * 3 + 2];

    float d0 = xi0 - xj0, d1 = xi1 - xj1, d2 = xi2 - xj2;
    float dsq = d0 * d0 + d1 * d1 + d2 * d2;
    float e0 = yi0 - yj0, e1 = yi1 - yj1, e2 = yi2 - yj2;
    float sq0 = e0 * e0 + e1 * e1 + e2 * e2;
    float r0v = (sq0 > 0.f) ? sqrtf(sq0) : 0.f;
    mySc[32 + p] = dsq;  // stash per-pair dsq (both lane halves write same value)

    // ---- EDGE branch: layer 1 (decomposed) -> silu -> f16 tile in LDS
#pragma unroll 4
    for (int u = 0; u < 64; ++u) {
      int f = fhalf + u;
      float v = AeRow[f] + BeJ[f] + dsq * wdE[f] + r0v * wrE[f] + bE[f];
      myM1[p * FF + f] = (_Float16)siluf(v);
    }
    // layer 2: 16x128 @ 128x128^T via 8 N-tiles x 4 K-steps of WMMA
    v16h aF0 = load_a_frag(myM1 + 0, FF);
    v16h aF1 = load_a_frag(myM1 + 32, FF);
    v16h aF2 = load_a_frag(myM1 + 64, FF);
    v16h aF3 = load_a_frag(myM1 + 96, FF);
#pragma unroll
    for (int nt = 0; nt < 8; ++nt) {
      v8f acc = {};
      acc = __builtin_amdgcn_wmma_f32_16x16x32_f16(false, aF0, false, load_b_frag(w2e, nt * 16, 0),  (short)0, acc, false, false);
      acc = __builtin_amdgcn_wmma_f32_16x16x32_f16(false, aF1, false, load_b_frag(w2e, nt * 16, 32), (short)0, acc, false, false);
      acc = __builtin_amdgcn_wmma_f32_16x16x32_f16(false, aF2, false, load_b_frag(w2e, nt * 16, 64), (short)0, acc, false, false);
      acc = __builtin_amdgcn_wmma_f32_16x16x32_f16(false, aF3, false, load_b_frag(w2e, nt * 16, 96), (short)0, acc, false, false);
      float bb = b2E[nt * 16 + nsel];
#pragma unroll
      for (int r = 0; r < 8; ++r)
        myO2[(r + 8 * hi) * FF + nt * 16 + nsel] = siluf(acc[r] + bb);
    }
    // gate e per pair: full-wave half-dots combined with shfl_xor(16)
    {
      float part = 0.f;
#pragma unroll 8
      for (int u = 0; u < 64; ++u) part += myO2[p * FF + fhalf + u] * wiE[fhalf + u];
      part += __shfl_xor(part, 16, 32);
      float e = (jb + p == i) ? 0.f : sigmf(part + eib);
      mySc[p] = e;  // both halves write identical value
    }
    float ev[16];
#pragma unroll
    for (int q = 0; q < 16; ++q) ev[q] = mySc[q];
    {
      float aggAcc[4];
#pragma unroll
      for (int u = 0; u < 4; ++u) {
        int f = lane * 4 + u;
        float s = 0.f;
#pragma unroll
        for (int q = 0; q < 16; ++q) s += ev[q] * myO2[q * FF + f];
        aggAcc[u] = s;
      }
#pragma unroll
      for (int u = 0; u < 4; ++u)
        atomicAdd(&aggG[(b * NN + i) * FF + lane * 4 + u], aggAcc[u]);
    }

    // ---- COORDINATE branch (same pipeline, cor weights)
#pragma unroll 4
    for (int u = 0; u < 64; ++u) {
      int f = fhalf + u;
      float v = AcRow[f] + BcJ[f] + dsq * wdC[f] + r0v * wrC[f] + bC[f];
      myM1[p * FF + f] = (_Float16)siluf(v);
    }
    aF0 = load_a_frag(myM1 + 0, FF);
    aF1 = load_a_frag(myM1 + 32, FF);
    aF2 = load_a_frag(myM1 + 64, FF);
    aF3 = load_a_frag(myM1 + 96, FF);
#pragma unroll
    for (int nt = 0; nt < 8; ++nt) {
      v8f acc = {};
      acc = __builtin_amdgcn_wmma_f32_16x16x32_f16(false, aF0, false, load_b_frag(w2c, nt * 16, 0),  (short)0, acc, false, false);
      acc = __builtin_amdgcn_wmma_f32_16x16x32_f16(false, aF1, false, load_b_frag(w2c, nt * 16, 32), (short)0, acc, false, false);
      acc = __builtin_amdgcn_wmma_f32_16x16x32_f16(false, aF2, false, load_b_frag(w2c, nt * 16, 64), (short)0, acc, false, false);
      acc = __builtin_amdgcn_wmma_f32_16x16x32_f16(false, aF3, false, load_b_frag(w2c, nt * 16, 96), (short)0, acc, false, false);
      float bb = b2C[nt * 16 + nsel];
#pragma unroll
      for (int r = 0; r < 8; ++r)
        myO2[(r + 8 * hi) * FF + nt * 16 + nsel] = siluf(acc[r] + bb);
    }
    // cw per pair (full-wave dot), then coordinate shift contribution
    {
      float part = 0.f;
#pragma unroll 8
      for (int u = 0; u < 64; ++u) part += myO2[p * FF + fhalf + u] * w3C[fhalf + u];
      part += __shfl_xor(part, 16, 32);
      mySc[p] = part + c3b;  // cw, both halves identical
    }
    if (lane < 3) {
      int dd = lane;
      float xid = (dd == 0) ? xi0 : ((dd == 1) ? xi1 : xi2);
      float s = 0.f;
#pragma unroll
      for (int q = 0; q < 16; ++q) {
        int j2 = jb + q;
        float dq = mySc[32 + q];
        float rr = (dq > 0.f) ? sqrtf(dq) : 0.f;
        float diffd = xid - x[(b * NN + j2) * 3 + dd];
        s += mySc[q] * diffd * fastrcp(rr + 1.0f);
      }
      atomicAdd(&xsumG[(b * NN + i) * 3 + dd], s);
    }
  }  // it
}

// ---------------------------------------------------------------------------
// Stage 3: node MLP  h_next = h_ln + W2 @ silu(W1 @ [h_ln, agg] + b1) + b2
// plus x_next = x_ln + xsum.
__global__ void __launch_bounds__(128) stage3_node(
    const float* __restrict__ hln, const float* __restrict__ agg,
    const float* __restrict__ xln, const float* __restrict__ xsumG,
    const float* __restrict__ n1w, const float* __restrict__ n1b,
    const float* __restrict__ n2w, const float* __restrict__ n2b,
    float* __restrict__ xnext, float* __restrict__ hnext) {
  int row = blockIdx.x;
  int t = threadIdx.x;
  __shared__ float sh[FF], sa[FF], nu[FF];
  sh[t] = hln[row * FF + t];
  sa[t] = agg[row * FF + t];
  if (t < 3) xnext[row * 3 + t] = xln[row * 3 + t] + xsumG[row * 3 + t];
  __syncthreads();
  float s = n1b[t];
  const float* w1 = n1w + t * (2 * FF);
#pragma unroll 8
  for (int k = 0; k < FF; ++k) s += sh[k] * w1[k];
#pragma unroll 8
  for (int k = 0; k < FF; ++k) s += sa[k] * w1[FF + k];
  nu[t] = siluf(s);
  __syncthreads();
  float s2 = n2b[t];
  const float* w2 = n2w + t * FF;
#pragma unroll 8
  for (int k = 0; k < FF; ++k) s2 += nu[k] * w2[k];
  hnext[row * FF + t] = sh[t] + s2;
}

// ---------------------------------------------------------------------------
extern "C" void kernel_launch(void* const* d_in, const int* in_sizes, int n_in,
                              void* d_out, int out_size, void* d_ws, size_t ws_size,
                              hipStream_t stream) {
  (void)in_sizes; (void)n_in; (void)out_size; (void)ws_size;
  const float* x       = (const float*)d_in[0];
  const float* h       = (const float*)d_in[1];
  const float* x0      = (const float*)d_in[2];
  const float* lnh_w   = (const float*)d_in[3];
  const float* lnh_b   = (const float*)d_in[4];
  const float* lnx_w   = (const float*)d_in[5];
  const float* lnx_b   = (const float*)d_in[6];
  const float* edg1_w  = (const float*)d_in[7];
  const float* edg1_b  = (const float*)d_in[8];
  const float* edg2_w  = (const float*)d_in[9];
  const float* edg2_b  = (const float*)d_in[10];
  const float* edgi_w  = (const float*)d_in[11];
  const float* edgi_b  = (const float*)d_in[12];
  const float* node1_w = (const float*)d_in[13];
  const float* node1_b = (const float*)d_in[14];
  const float* node2_w = (const float*)d_in[15];
  const float* node2_b = (const float*)d_in[16];
  const float* cor1_w  = (const float*)d_in[17];
  const float* cor1_b  = (const float*)d_in[18];
  const float* cor2_w  = (const float*)d_in[19];
  const float* cor2_b  = (const float*)d_in[20];
  const float* cor3_w  = (const float*)d_in[21];
  const float* cor3_b  = (const float*)d_in[22];

  float* ws  = (float*)d_ws;
  float* hln = ws;                    // B*N*F
  float* xln = hln + BB * NN * FF;    // B*N*3
  float* Ae  = xln + BB * NN * 3;     // B*N*F
  float* Be  = Ae + BB * NN * FF;
  float* Ac  = Be + BB * NN * FF;
  float* Bc  = Ac + BB * NN * FF;
  float* aggG  = Bc + BB * NN * FF;   // B*N*F
  float* xsumG = aggG + BB * NN * FF; // B*N*3
  _Float16* w16 = (_Float16*)(xsumG + BB * NN * 3);  // 2*F*F halves (16B-aligned)

  float* xnext = (float*)d_out;              // B*N*3
  float* hnext = xnext + BB * NN * 3;        // B*N*F

  hipFuncSetAttribute((const void*)stage2_pairs,
                      hipFuncAttributeMaxDynamicSharedMemorySize, SMEM_BYTES);

  stage0_cvt_weights<<<(2 * FF * FF) / 256, 256, 0, stream>>>(edg2_w, cor2_w, w16);
  stage1_ln_proj<<<BB * NN, 128, 0, stream>>>(x, h, lnh_w, lnh_b, lnx_w, lnx_b,
                                              edg1_w, cor1_w, hln, xln, Ae, Be, Ac, Bc,
                                              aggG, xsumG);
  stage2_pairs<<<BB * 16 * 4, 128, SMEM_BYTES, stream>>>(
      x, x0, Ae, Be, Ac, Bc, w16,
      edg1_w, edg1_b, edg2_b, edgi_w, edgi_b,
      cor1_w, cor1_b, cor2_b, cor3_w, cor3_b,
      aggG, xsumG);
  stage3_node<<<BB * NN, 128, 0, stream>>>(hln, aggG, xln, xsumG,
                                           node1_w, node1_b, node2_w, node2_b,
                                           xnext, hnext);
}
